// SimpleRNN_9277129360056
// MI455X (gfx1250) — matrix operand order repaired and checked by reading, not compiled, not run on hardware
//
#include <hip/hip_runtime.h>
#include <hip/hip_bf16.h>
#include <math.h>

// ---------------------------------------------------------------------------
// 2-layer tanh RNN on MI455X (gfx1250, wave32, WMMA).
// B=128,T=1024,I=256,H=512,O=256.
// Batch rows are independent -> 8 workgroups, each owns a 16-row batch tile
// and runs the full T=1024 recurrence with h resident in LDS (bf16).
// Layer 1 uses the SAME tensor as input and hidden (reference aliases them),
// so W_ih1+W_hh1 are folded into one bf16 matrix in a prep pass.
// Matrix ops: v_wmma_f32_16x16x32_bf16, f32 accumulation, f32 tanh.
// ---------------------------------------------------------------------------

#define B_ 128
#define T_ 1024
#define I_ 256
#define H_ 512
#define O_ 256

#define XLD 264   // 256 + 8 pad (16B) -> conflict-free ds_load_b128
#define HLD 520   // 512 + 8 pad

typedef __attribute__((ext_vector_type(16))) __bf16 v16bf;
typedef __attribute__((ext_vector_type(8)))  __bf16 v8bf;
typedef __attribute__((ext_vector_type(8)))  float  v8f;

union Frag16 { v16bf v; v8bf h[2]; };

// A-fragment (16x32 bf16, M=batch rows) from LDS, per ISA 16-bit A layout:
// lanes 0-15: M=lane, K = k0+{0..7} and k0+{16..23}
// lanes 16-31: M=lane-16, K = k0+{8..15} and k0+{24..31}
__device__ __forceinline__ v16bf load_a_frag(const __bf16* sA, int lda, int k0, int lane) {
    const int m   = lane & 15;
    const int kof = (lane >> 4) * 8;
    const __bf16* p = sA + m * lda + k0 + kof;
    Frag16 f;
    f.h[0] = *(const v8bf*)(p);
    f.h[1] = *(const v8bf*)(p + 16);
    return f.v;
}

// B-fragment (32x16 bf16): B[k][n] = W[row0+n][k] (row-major weight, we
// multiply by W.T so no transpose needed). Per ISA 16-bit B layout:
// lanes 0-15: N=lane, K=k0+{0..7},{16..23} ; lanes 16-31: N=lane-16, K=k0+{8..15},{24..31}.
// two 16B global reads per lane (L2-resident).
__device__ __forceinline__ v16bf load_b_frag(const __bf16* __restrict__ gB, int ldb,
                                             int row0, int k0, int lane) {
    const int n  = lane & 15;
    const int kb = (lane >> 4) * 8;
    const __bf16* p = gB + (row0 + n) * ldb + k0 + kb;
    Frag16 f;
    f.h[0] = *(const v8bf*)(p);
    f.h[1] = *(const v8bf*)(p + 16);
    return f.v;
}

__global__ __launch_bounds__(256, 1) void rnn_persistent_kernel(
    const float* __restrict__ x,      // (B,T,I) f32
    const float* __restrict__ h0,     // (B,H)   f32
    const __bf16* __restrict__ Wih0b, // (H,I)   bf16
    const __bf16* __restrict__ Whh0b, // (H,H)   bf16
    const __bf16* __restrict__ W1c,   // (H,H)   bf16  = Wih1+Whh1
    const __bf16* __restrict__ Woutb, // (O,H)   bf16
    const float* __restrict__ b0,     // (H) = b_ih0+b_hh0
    const float* __restrict__ b1,     // (H) = b_ih1+b_hh1
    const float* __restrict__ bo,     // (O) = b_out
    float* __restrict__ y)            // (B,T,O) f32
{
    __shared__ __bf16 sx [16 * XLD];
    __shared__ __bf16 sh [16 * HLD];
    __shared__ __bf16 sh2[16 * HLD];

    const int tid  = threadIdx.x;
    const int wave = tid >> 5;
    const int lane = tid & 31;
    const int bt   = blockIdx.x;            // batch tile (16 rows)

    const int n_  = lane & 15;
    const int mb  = (lane >> 4) * 8;        // C/D: VGPR r -> row mb+r

    // ---- init h from h0 ----
    for (int e = tid; e < 16 * H_; e += 256) {
        const int r = e >> 9, c = e & (H_ - 1);
        sh[r * HLD + c] = (__bf16)h0[(bt * 16 + r) * H_ + c];
    }
    __syncthreads();

    // per-thread x-tile slice: row = tid/16, 16 contiguous cols
    const int xr  = tid >> 4;
    const int xc0 = (tid & 15) * 16;
    const float* xrow = x + ((size_t)(bt * 16 + xr) * T_) * I_ + xc0;

    for (int t = 0; t < T_; ++t) {
        // ---- stage x_t tile (f32 -> bf16) into LDS ----
        {
            const float* px = xrow + (size_t)t * I_;
            v8bf o0, o1;
            #pragma unroll
            for (int u = 0; u < 8; ++u) o0[u] = (__bf16)px[u];
            #pragma unroll
            for (int u = 0; u < 8; ++u) o1[u] = (__bf16)px[8 + u];
            *(v8bf*)(&sx[xr * XLD + xc0])     = o0;
            *(v8bf*)(&sx[xr * XLD + xc0 + 8]) = o1;
            if (t + 1 < T_)  // overlap next step's x fetch with the GEMM chain
                __builtin_prefetch(xrow + (size_t)(t + 1) * I_, 0, 0);
        }
        __syncthreads();

        // ---- layer 0: h2 = tanh(x@Wih0.T + h@Whh0.T + b0), 4 N-tiles/wave ----
        {
            v8f acc[4];
            #pragma unroll
            for (int i = 0; i < 4; ++i) {
                const float bv = b0[(wave * 4 + i) * 16 + n_];
                #pragma unroll
                for (int r = 0; r < 8; ++r) acc[i][r] = bv;
            }
            for (int k0 = 0; k0 < I_; k0 += 32) {
                v16bf a = load_a_frag(sx, XLD, k0, lane);
                #pragma unroll
                for (int i = 0; i < 4; ++i) {
                    v16bf b = load_b_frag(Wih0b, I_, (wave * 4 + i) * 16, k0, lane);
                    acc[i] = __builtin_amdgcn_wmma_f32_16x16x32_bf16(
                        false, a, false, b, (short)0, acc[i], false, false);
                }
            }
            for (int k0 = 0; k0 < H_; k0 += 32) {
                v16bf a = load_a_frag(sh, HLD, k0, lane);
                #pragma unroll
                for (int i = 0; i < 4; ++i) {
                    v16bf b = load_b_frag(Whh0b, H_, (wave * 4 + i) * 16, k0, lane);
                    acc[i] = __builtin_amdgcn_wmma_f32_16x16x32_bf16(
                        false, a, false, b, (short)0, acc[i], false, false);
                }
            }
            #pragma unroll
            for (int i = 0; i < 4; ++i) {
                const int j = (wave * 4 + i) * 16 + n_;
                #pragma unroll
                for (int r = 0; r < 8; ++r)
                    sh2[(mb + r) * HLD + j] = (__bf16)tanhf(acc[i][r]);
            }
        }
        __syncthreads();

        // ---- layer 1: h = tanh(h2@(Wih1+Whh1).T + b1)  (input == hidden) ----
        {
            v8f acc[4];
            #pragma unroll
            for (int i = 0; i < 4; ++i) {
                const float bv = b1[(wave * 4 + i) * 16 + n_];
                #pragma unroll
                for (int r = 0; r < 8; ++r) acc[i][r] = bv;
            }
            for (int k0 = 0; k0 < H_; k0 += 32) {
                v16bf a = load_a_frag(sh2, HLD, k0, lane);
                #pragma unroll
                for (int i = 0; i < 4; ++i) {
                    v16bf b = load_b_frag(W1c, H_, (wave * 4 + i) * 16, k0, lane);
                    acc[i] = __builtin_amdgcn_wmma_f32_16x16x32_bf16(
                        false, a, false, b, (short)0, acc[i], false, false);
                }
            }
            #pragma unroll
            for (int i = 0; i < 4; ++i) {
                const int j = (wave * 4 + i) * 16 + n_;
                #pragma unroll
                for (int r = 0; r < 8; ++r)
                    sh[(mb + r) * HLD + j] = (__bf16)tanhf(acc[i][r]);
            }
        }
        __syncthreads();

        // ---- output head: y_t = h@W_out.T + b_out, 2 N-tiles/wave ----
        {
            v8f acc[2];
            #pragma unroll
            for (int i = 0; i < 2; ++i) {
                const float bv = bo[(wave * 2 + i) * 16 + n_];
                #pragma unroll
                for (int r = 0; r < 8; ++r) acc[i][r] = bv;
            }
            for (int k0 = 0; k0 < H_; k0 += 32) {
                v16bf a = load_a_frag(sh, HLD, k0, lane);
                #pragma unroll
                for (int i = 0; i < 2; ++i) {
                    v16bf b = load_b_frag(Woutb, H_, (wave * 2 + i) * 16, k0, lane);
                    acc[i] = __builtin_amdgcn_wmma_f32_16x16x32_bf16(
                        false, a, false, b, (short)0, acc[i], false, false);
                }
            }
            #pragma unroll
            for (int i = 0; i < 2; ++i) {
                const int o = (wave * 2 + i) * 16 + n_;
                #pragma unroll
                for (int r = 0; r < 8; ++r)
                    y[((size_t)(bt * 16 + mb + r) * T_ + t) * O_ + o] = acc[i][r];
            }
        }
        // no barrier needed: next-iter sx write is covered by this iter's
        // barriers; head only reads sh (rewritten 2 barriers into next iter).
    }
}

// ---------------- prep kernels (weights -> bf16, fold layer-1) -------------
__global__ void cvt_bf16_kernel(const float* __restrict__ s, __bf16* __restrict__ d, int n) {
    int i = blockIdx.x * blockDim.x + threadIdx.x;
    if (i < n) d[i] = (__bf16)s[i];
}
__global__ void addcvt_bf16_kernel(const float* __restrict__ a, const float* __restrict__ b,
                                   __bf16* __restrict__ d, int n) {
    int i = blockIdx.x * blockDim.x + threadIdx.x;
    if (i < n) d[i] = (__bf16)(a[i] + b[i]);
}
__global__ void addf32_kernel(const float* __restrict__ a, const float* __restrict__ b,
                              float* __restrict__ d, int n) {
    int i = blockIdx.x * blockDim.x + threadIdx.x;
    if (i < n) d[i] = a[i] + b[i];
}

extern "C" void kernel_launch(void* const* d_in, const int* in_sizes, int n_in,
                              void* d_out, int out_size, void* d_ws, size_t ws_size,
                              hipStream_t stream) {
    const float* x     = (const float*)d_in[0];
    const float* h0    = (const float*)d_in[1];
    const float* Wih0  = (const float*)d_in[2];
    const float* b_ih0 = (const float*)d_in[3];
    const float* Whh0  = (const float*)d_in[4];
    const float* b_hh0 = (const float*)d_in[5];
    const float* Wih1  = (const float*)d_in[6];
    const float* b_ih1 = (const float*)d_in[7];
    const float* Whh1  = (const float*)d_in[8];
    const float* b_hh1 = (const float*)d_in[9];
    const float* W_out = (const float*)d_in[10];
    const float* b_out = (const float*)d_in[11];

    char* ws = (char*)d_ws;
    size_t off = 0;
    __bf16* Wih0b = (__bf16*)(ws + off); off += (size_t)H_ * I_ * 2;   // 256 KB
    __bf16* Whh0b = (__bf16*)(ws + off); off += (size_t)H_ * H_ * 2;   // 512 KB
    __bf16* W1c   = (__bf16*)(ws + off); off += (size_t)H_ * H_ * 2;   // 512 KB
    __bf16* Woutb = (__bf16*)(ws + off); off += (size_t)O_ * H_ * 2;   // 256 KB
    float*  b0    = (float*)(ws + off);  off += (size_t)H_ * 4;
    float*  b1    = (float*)(ws + off);  off += (size_t)H_ * 4;

    const int TB = 256;
    cvt_bf16_kernel   <<<(H_ * I_ + TB - 1) / TB, TB, 0, stream>>>(Wih0, Wih0b, H_ * I_);
    cvt_bf16_kernel   <<<(H_ * H_ + TB - 1) / TB, TB, 0, stream>>>(Whh0, Whh0b, H_ * H_);
    addcvt_bf16_kernel<<<(H_ * H_ + TB - 1) / TB, TB, 0, stream>>>(Wih1, Whh1, W1c, H_ * H_);
    cvt_bf16_kernel   <<<(O_ * H_ + TB - 1) / TB, TB, 0, stream>>>(W_out, Woutb, O_ * H_);
    addf32_kernel     <<<(H_ + TB - 1) / TB, TB, 0, stream>>>(b_ih0, b_hh0, b0, H_);
    addf32_kernel     <<<(H_ + TB - 1) / TB, TB, 0, stream>>>(b_ih1, b_hh1, b1, H_);

    rnn_persistent_kernel<<<B_ / 16, 256, 0, stream>>>(
        x, h0, Wih0b, Whh0b, W1c, Woutb, b0, b1, b_out, (float*)d_out);
}
